// tmGCN_90340342104100
// MI455X (gfx1250) — compile-verified, matrix operand order checked by reading, not run on hardware
//
#include <hip/hip_runtime.h>
#include <hip/hip_bf16.h>

// Problem constants (match reference)
#define NNODE 8192
#define INDIM 343
#define HID   1000
#define HIDP  1024   // HID padded to multiple of 128
#define OUTD  4
#define KPAD1 352    // INDIM padded to multiple of 32
#define EPSV  1e-5f

typedef __attribute__((ext_vector_type(16))) __bf16 bf16x16;
typedef __attribute__((ext_vector_type(8)))  float  f32x8;

#define LDSW 40   // LDS row stride in bf16 elements: 32 + 8 pad (conflict-free frags)

// ---- gfx1250 async global->LDS path (graceful fallback if builtin absent) --
#if defined(__has_builtin)
#if __has_builtin(__builtin_amdgcn_global_load_async_to_lds_b128) && \
    __has_builtin(__builtin_amdgcn_s_wait_asynccnt)
#define USE_ASYNC_LDS 1
#endif
#endif
#ifndef USE_ASYNC_LDS
#define USE_ASYNC_LDS 0
#endif

#if USE_ASYNC_LDS
typedef __attribute__((ext_vector_type(4))) int v4i;
typedef __attribute__((address_space(1))) void gvoid;
typedef __attribute__((address_space(3))) void lvoid;
typedef __attribute__((address_space(1))) v4i gv4i;
typedef __attribute__((address_space(3))) v4i lv4i;
#endif

// Copy 16 bf16 (32 B) from global to LDS.
__device__ __forceinline__ void copy_row16(const __bf16* src, __bf16* dst) {
#if USE_ASYNC_LDS
  // builtin signature (from clang diagnostic): param0 = AS1 int4* (non-const),
  // then LDS dst, imm offset, imm cpol.
  gv4i* g = (gv4i*)(gvoid*)(void*)const_cast<__bf16*>(src);
  lv4i* l = (lv4i*)(lvoid*)(void*)dst;
  __builtin_amdgcn_global_load_async_to_lds_b128(g, l, 0, 0);
  gv4i* g2 = (gv4i*)(gvoid*)(void*)const_cast<__bf16*>(src + 8);
  lv4i* l2 = (lv4i*)(lvoid*)(void*)(dst + 8);
  __builtin_amdgcn_global_load_async_to_lds_b128(g2, l2, 0, 0);
#else
  uint4 v0 = ((const uint4*)src)[0];
  uint4 v1 = ((const uint4*)src)[1];
  ((uint4*)dst)[0] = v0;
  ((uint4*)dst)[1] = v1;
#endif
}

__device__ __forceinline__ void wait_stage() {
#if USE_ASYNC_LDS
  __builtin_amdgcn_s_wait_asynccnt(0);  // my async copies into the buffer we read next
#endif
}

// Build a 16x32 bf16 WMMA operand fragment from an LDS tile (row stride LDSW).
// Per CDNA5 ISA 16-bit A 16x32 layout: lane L -> row base+(L&15), khalf=L>>4;
// two 16-byte reads cover K {kh*8..+7} and {16+kh*8..+7}. B fragments (stored
// transposed, N x K) use the identical layout.
__device__ __forceinline__ bf16x16 load_frag(const __bf16* s, int row, int khalf) {
  union { bf16x16 v; uint4 q[2]; } u;
  u.q[0] = *(const uint4*)(s + row * LDSW + khalf * 8);
  u.q[1] = *(const uint4*)(s + row * LDSW + 16 + khalf * 8);
  return u.v;
}

// ---------------------------------------------------------------------------
// prep: W1 [343 x 1000] f32 -> W1T [1024 x 352] bf16, zero padded (transposed)
// ---------------------------------------------------------------------------
__global__ __launch_bounds__(256) void prep_w1t(const float* __restrict__ W1,
                                                __bf16* __restrict__ W1T) {
  int idx = blockIdx.x * 256 + threadIdx.x;
  if (idx >= HIDP * KPAD1) return;
  int n = idx / KPAD1;
  int k = idx % KPAD1;
  float v = (n < HID && k < INDIM) ? W1[(size_t)k * HID + n] : 0.f;
  W1T[idx] = (__bf16)v;
}

// ---------------------------------------------------------------------------
// prep: features [8192 x 343] f32 -> featP [8192 x 352] bf16, zero padded
// ---------------------------------------------------------------------------
__global__ __launch_bounds__(256) void prep_feat(const float* __restrict__ X,
                                                 __bf16* __restrict__ XP) {
  int idx = blockIdx.x * 256 + threadIdx.x;
  if (idx >= NNODE * KPAD1) return;
  int r = idx / KPAD1;
  int k = idx % KPAD1;
  float v = (k < INDIM) ? X[(size_t)r * INDIM + k] : 0.f;
  XP[idx] = (__bf16)v;
}

// ---------------------------------------------------------------------------
// 128x128 block GEMM, K step 32, bf16 WMMA / f32 accumulate, LDS ping-pong.
//   AF32=true : A is f32 row-major (adj), converted to bf16 while staging
//   AF32=false: A is bf16 row-major, pre-padded (featP)
//   B: bf16 stored transposed [N x ldb]
// MODE 0: store D transposed bf16 -> CtrBf[HIDP x NNODE]   (packed b128 stores)
// MODE 1: D = relu(D + bias[n]),  store f32 -> Crow[NNODE x HIDP]
// ---------------------------------------------------------------------------
template <int MODE, bool AF32>
__global__ __launch_bounds__(256)
void gemm_bf16_128x128(const float* __restrict__ Af, const __bf16* __restrict__ Ah,
                       int lda, int K,
                       const __bf16* __restrict__ B, int ldb,
                       const float* __restrict__ bias,
                       float* __restrict__ Crow,
                       __bf16* __restrict__ CtrBf) {
  __shared__ __bf16 As[2][128 * LDSW];
  __shared__ __bf16 Bs[2][128 * LDSW];

  const int tid   = threadIdx.x;
  const int wave  = tid >> 5;
  const int lane  = tid & 31;
  const int wm    = (wave & 3) * 32;
  const int wn    = (wave >> 2) * 64;
  const int lr    = lane & 15;
  const int kh    = lane >> 4;
  const int mBase = blockIdx.y * 128;
  const int nBase = blockIdx.x * 128;

  const int srow  = tid >> 1;  // staging row/col 0..127
  const int shalf = tid & 1;   // staging K half (16 elements each)

  f32x8 acc[2][4] = {};

  auto stageA = [&](int buf, int k0) {
    __bf16* dst = &As[buf][srow * LDSW + shalf * 16];
    if (AF32) {
      const float* src = Af + (size_t)(mBase + srow) * lda + k0 + shalf * 16;
      __bf16 tmp[16] __attribute__((aligned(16)));
#pragma unroll
      for (int c4 = 0; c4 < 4; ++c4) {
        float4 f = ((const float4*)src)[c4];
        tmp[c4 * 4 + 0] = (__bf16)f.x;
        tmp[c4 * 4 + 1] = (__bf16)f.y;
        tmp[c4 * 4 + 2] = (__bf16)f.z;
        tmp[c4 * 4 + 3] = (__bf16)f.w;
      }
      ((uint4*)dst)[0] = ((const uint4*)tmp)[0];
      *(uint4*)(dst + 8) = ((const uint4*)tmp)[1];
      if (k0 + 32 < K) __builtin_prefetch(src + 32, 0, 1);  // next K tile of adj
    } else {
      copy_row16(Ah + (size_t)(mBase + srow) * lda + k0 + shalf * 16, dst);
    }
  };
  auto stageB = [&](int buf, int k0) {
    copy_row16(B + (size_t)(nBase + srow) * ldb + k0 + shalf * 16,
               &Bs[buf][srow * LDSW + shalf * 16]);
  };

  stageA(0, 0);
  stageB(0, 0);

  const int nsteps = K >> 5;
  for (int it = 0; it < nsteps; ++it) {
    const int buf = it & 1;
    wait_stage();
    __syncthreads();  // staging of `buf` visible; prior reads of buf^1 finished
    if (it + 1 < nsteps) {
      stageA(buf ^ 1, (it + 1) << 5);
      stageB(buf ^ 1, (it + 1) << 5);
    }

    const __bf16* as = As[buf];
    const __bf16* bs = Bs[buf];
    bf16x16 af[2];
#pragma unroll
    for (int i = 0; i < 2; ++i) af[i] = load_frag(as, wm + 16 * i + lr, kh);
    bf16x16 bfm[4];
#pragma unroll
    for (int j = 0; j < 4; ++j) bfm[j] = load_frag(bs, wn + 16 * j + lr, kh);

#pragma unroll
    for (int i = 0; i < 2; ++i)
#pragma unroll
      for (int j = 0; j < 4; ++j)
        acc[i][j] = __builtin_amdgcn_wmma_f32_16x16x32_bf16(
            false, af[i], false, bfm[j], (short)0, acc[i][j], false, false);
  }

  // ---- epilogue; D layout: VGPR vr, lane -> M = vr + 8*(lane>>4), N = lane&15
#pragma unroll
  for (int i = 0; i < 2; ++i) {
#pragma unroll
    for (int j = 0; j < 4; ++j) {
      const int gn = nBase + wn + 16 * j + lr;
      if (MODE == 1) {
        const float bv = (gn < HID) ? bias[gn] : 0.f;
#pragma unroll
        for (int vr = 0; vr < 8; ++vr) {
          const int gm = mBase + wm + 16 * i + vr + 8 * kh;
          float v = acc[i][j][vr] + bv;
          Crow[(size_t)gm * HIDP + gn] = fmaxf(v, 0.f);
        }
      } else {
        // 8 consecutive gm values -> one packed 16-byte store
        union { uint4 q; __bf16 h[8]; } pk;
#pragma unroll
        for (int vr = 0; vr < 8; ++vr) pk.h[vr] = (__bf16)acc[i][j][vr];
        const int gm0 = mBase + wm + 16 * i + 8 * kh;
        *(uint4*)(CtrBf + (size_t)gn * NNODE + gm0) = pk.q;
      }
    }
  }
}

// ---------------------------------------------------------------------------
// LayerNorm over HID + 4-wide linear head. One 256-thread block per row.
// ---------------------------------------------------------------------------
__global__ __launch_bounds__(256)
void ln_head(const float* __restrict__ h1, const float* __restrict__ gamma,
             const float* __restrict__ beta, const float* __restrict__ Wm,
             const float* __restrict__ bm, float* __restrict__ out) {
  __shared__ float2 r2[256];
  __shared__ float4 r4[256];
  const int row = blockIdx.x;
  const int tid = threadIdx.x;
  const float* x = h1 + (size_t)row * HIDP;

  float s = 0.f, ss = 0.f;
  for (int c = tid; c < HID; c += 256) {
    float v = x[c];
    s += v;
    ss += v * v;
  }
  r2[tid] = make_float2(s, ss);
  __syncthreads();
  for (int off = 128; off > 0; off >>= 1) {
    if (tid < off) {
      r2[tid].x += r2[tid + off].x;
      r2[tid].y += r2[tid + off].y;
    }
    __syncthreads();
  }
  const float mu   = r2[0].x * (1.f / HID);
  const float var  = r2[0].y * (1.f / HID) - mu * mu;
  const float rstd = rsqrtf(var + EPSV);

  float4 p = make_float4(0.f, 0.f, 0.f, 0.f);
  for (int c = tid; c < HID; c += 256) {
    float xn = (x[c] - mu) * rstd * gamma[c] + beta[c];
    float4 w = ((const float4*)Wm)[c];
    p.x += xn * w.x;
    p.y += xn * w.y;
    p.z += xn * w.z;
    p.w += xn * w.w;
  }
  r4[tid] = p;
  __syncthreads();
  for (int off = 128; off > 0; off >>= 1) {
    if (tid < off) {
      r4[tid].x += r4[tid + off].x;
      r4[tid].y += r4[tid + off].y;
      r4[tid].z += r4[tid + off].z;
      r4[tid].w += r4[tid + off].w;
    }
    __syncthreads();
  }
  if (tid == 0) {
    float4 r = r4[0];
    out[(size_t)row * OUTD + 0] = r.x + bm[0];
    out[(size_t)row * OUTD + 1] = r.y + bm[1];
    out[(size_t)row * OUTD + 2] = r.z + bm[2];
    out[(size_t)row * OUTD + 3] = r.w + bm[3];
  }
}

// ---------------------------------------------------------------------------
extern "C" void kernel_launch(void* const* d_in, const int* in_sizes, int n_in,
                              void* d_out, int out_size, void* d_ws, size_t ws_size,
                              hipStream_t stream) {
  const float* adj  = (const float*)d_in[0];
  const float* feat = (const float*)d_in[1];
  const float* W1   = (const float*)d_in[2];
  const float* b1   = (const float*)d_in[3];
  const float* gam  = (const float*)d_in[4];
  const float* bet  = (const float*)d_in[5];
  const float* Wm   = (const float*)d_in[6];
  const float* bm   = (const float*)d_in[7];
  float* out = (float*)d_out;

  // Workspace layout (fully written each call):
  //   xwT   : [HIDP x NNODE] bf16 = 16 MB (L2-resident B operand of big GEMM)
  //   W1T   : [HIDP x KPAD1] bf16 ~ 0.7 MB
  //   featP : [NNODE x KPAD1] bf16 ~ 5.8 MB
  //   h1    : [NNODE x HIDP] f32  = 32 MB
  char* ws = (char*)d_ws;
  size_t off = 0;
  __bf16* xwT = (__bf16*)(ws + off);
  off += (size_t)HIDP * NNODE * 2;
  __bf16* W1T = (__bf16*)(ws + off);
  off += (size_t)HIDP * KPAD1 * 2;
  __bf16* featP = (__bf16*)(ws + off);
  off += (size_t)NNODE * KPAD1 * 2;
  float* h1 = (float*)(ws + off);

  prep_w1t<<<(HIDP * KPAD1 + 255) / 256, 256, 0, stream>>>(W1, W1T);
  prep_feat<<<(NNODE * KPAD1 + 255) / 256, 256, 0, stream>>>(feat, featP);

  // xw = X @ W1 (both operands pre-padded bf16), store transposed bf16
  gemm_bf16_128x128<0, false><<<dim3(HIDP / 128, NNODE / 128), 256, 0, stream>>>(
      nullptr, featP, KPAD1, KPAD1, W1T, KPAD1, nullptr, nullptr, xwT);

  // h1 = relu(adj @ xw + b1), adj converted f32->bf16 on the fly
  gemm_bf16_128x128<1, true><<<dim3(HIDP / 128, NNODE / 128), 256, 0, stream>>>(
      adj, nullptr, NNODE, NNODE, xwT, NNODE, b1, h1, nullptr);

  // LayerNorm + head
  ln_head<<<NNODE, 256, 0, stream>>>(h1, gam, bet, Wm, bm, out);
}
